// Encoder_61177514164477
// MI455X (gfx1250) — matrix operand order repaired and checked by reading, not compiled, not run on hardware
//
#include <hip/hip_runtime.h>
#include <hip/hip_bf16.h>

// ---------------------------------------------------------------------------
// Meta-GCN-LSTM for MI455X (gfx1250).  All heavy GEMMs use
// v_wmma_f32_16x16x32_bf16 (fp32 accumulation).  G (25MB bf16) lives in L2.
// ---------------------------------------------------------------------------

typedef __attribute__((ext_vector_type(16))) __bf16        v16bf;
typedef __attribute__((ext_vector_type(8)))  float         v8f;
typedef __attribute__((ext_vector_type(4)))  unsigned int  u32x4;

union FragAB { u32x4 q[2]; v16bf v; };

static constexpr int Lc = 2, Bc = 4, Tc = 8, Nc = 2048, Cc = 2, Hc = 64, Kc = 3, Mc = 32;
static constexpr int DOUTc = 4 * Hc;          // 256
static constexpr int DIN0  = Cc + Hc;         // 66
static constexpr int DIN1  = 2 * Hc;          // 128
static constexpr int DINP0 = 80;              // 66 padded to multiple of 16
static constexpr int DINP1 = 128;
static constexpr int KSUP0 = 256;             // 3*80=240 padded to multiple of 32
static constexpr int KSUP1 = 3 * DINP1;       // 384

__device__ __forceinline__ float sigf(float x) { return 1.0f / (1.0f + __expf(-x)); }
__device__ __forceinline__ float tanhfast(float x) { return 1.0f - 2.0f / (__expf(2.0f * x) + 1.0f); }

// ------------------------------- utility kernels ---------------------------

__global__ void k_cvt_bf16(const float* __restrict__ s, __bf16* __restrict__ d, long n) {
    long i  = (long)blockIdx.x * blockDim.x + threadIdx.x;
    long st = (long)gridDim.x * blockDim.x;
    for (; i < n; i += st) d[i] = (__bf16)s[i];
}

__global__ void k_zero32(unsigned int* __restrict__ p, long nwords) {
    long i  = (long)blockIdx.x * blockDim.x + threadIdx.x;
    long st = (long)gridDim.x * blockDim.x;
    for (; i < nwords; i += st) p[i] = 0u;
}

__global__ void k_copy_f32(const float* __restrict__ s, float* __restrict__ d, long n) {
    long i  = (long)blockIdx.x * blockDim.x + threadIdx.x;
    long st = (long)gridDim.x * blockDim.x;
    for (; i < n; i += st) d[i] = s[i];
}

// ----------------------- x_cat builders (transposed bf16) ------------------
// XcT layout: [B][dinp][N] bf16.  Pad rows (d >= din) are zeroed once up front.

__global__ void k_xcat0(const float* __restrict__ x_seq, const float* __restrict__ h0,
                        __bf16* __restrict__ XcT, int t) {
    int idx = blockIdx.x * blockDim.x + threadIdx.x;       // b*N + j
    if (idx >= Bc * Nc) return;
    int b = idx / Nc, j = idx % Nc;
    const float* xp = x_seq + (((size_t)b * Tc + t) * Nc + j) * Cc;
    const float* hp = h0 + ((size_t)b * Nc + j) * Hc;
    __bf16* o = XcT + (size_t)b * DINP0 * Nc + j;
    o[0]           = (__bf16)xp[0];
    o[(size_t)Nc]  = (__bf16)xp[1];
#pragma unroll
    for (int d = 0; d < Hc; ++d) o[(size_t)(Cc + d) * Nc] = (__bf16)hp[d];
}

__global__ void k_xcat1(const float* __restrict__ hs0_t, const float* __restrict__ h1,
                        __bf16* __restrict__ XcT) {
    int idx = blockIdx.x * blockDim.x + threadIdx.x;       // b*N + j
    if (idx >= Bc * Nc) return;
    int b = idx / Nc, j = idx % Nc;
    const float* ap = hs0_t + ((size_t)b * Nc + j) * Hc;
    const float* bp = h1    + ((size_t)b * Nc + j) * Hc;
    __bf16* o = XcT + (size_t)b * DINP1 * Nc + j;
#pragma unroll
    for (int d = 0; d < Hc; ++d) {
        o[(size_t)d * Nc]        = (__bf16)ap[d];
        o[(size_t)(Hc + d) * Nc] = (__bf16)bp[d];
    }
}

// ------------------------------ meta-MLP kernel ----------------------------
// Produces W transposed: WT[b][o][k*dinp+d] (bf16), and fp32 bias[b][o].
// Pad entries of WT are pre-zeroed and never touched here.

__global__ void k_meta(const float* __restrict__ xmeta, int t,
                       const float* __restrict__ lw1, const float* __restrict__ lb1,
                       const float* __restrict__ lw2, const float* __restrict__ lb2,
                       const float* __restrict__ bw1, const float* __restrict__ bb1,
                       const float* __restrict__ bw2, const float* __restrict__ bb2,
                       __bf16* __restrict__ WT, float* __restrict__ bias,
                       int din, int dinp, int ksup) {
    __shared__ float hid[64], hidb[64];
    int b   = blockIdx.x;
    int tid = threadIdx.x;
    const float* mp = xmeta + ((size_t)b * Tc + t) * Mc;
    if (tid < 64) {
        float a = lb1[tid];
        for (int m = 0; m < Mc; ++m) a += mp[m] * lw1[m * 64 + tid];
        hid[tid] = a > 0.f ? a : 0.f;
    } else if (tid < 128) {
        int i = tid - 64;
        float a = bb1[i];
        for (int m = 0; m < Mc; ++m) a += mp[m] * bw1[m * 64 + i];
        hidb[i] = a > 0.f ? a : 0.f;
    }
    __syncthreads();
    int outw   = Kc * din * DOUTc;
    int base   = blockIdx.y * blockDim.x + tid;
    int stride = gridDim.y * blockDim.x;
    for (int f = base; f < outw; f += stride) {
        float v = lb2[f];
#pragma unroll 8
        for (int r = 0; r < 64; ++r) v += hid[r] * lw2[(size_t)r * outw + f];
        int o  = f % DOUTc;
        int kd = f / DOUTc;
        int k  = kd / din, d = kd % din;
        WT[((size_t)b * DOUTc + o) * ksup + k * dinp + d] = (__bf16)v;
    }
    if (blockIdx.y == 0 && tid < DOUTc) {
        float v = bb2[tid];
        for (int r = 0; r < 64; ++r) v += hidb[r] * bw2[r * DOUTc + tid];
        bias[b * DOUTc + tid] = v;
    }
}

// ---------------------- GEMM1: sup = G[k] @ x_cat (WMMA) -------------------
// Output supB: [B][N][KSUP] bf16, columns k*DINP + d.
// Wave = 16 output rows x DINP cols; K loop over j in steps of 32.

template <int DINP, int KSUP>
__global__ __launch_bounds__(256) void k_gemm1(const __bf16* __restrict__ Gbf,
                                               const __bf16* __restrict__ XcT,
                                               __bf16* __restrict__ supB) {
    constexpr int NB = DINP / 16;
    int b    = blockIdx.z, k = blockIdx.y;
    int wave = threadIdx.x >> 5;
    int lane = threadIdx.x & 31;
    int half = lane >> 4, col = lane & 15;
    int m0   = blockIdx.x * 128 + wave * 16;

    // A (G[k]) lane row = m0+col; K chunks: [j0+half*8 .. +7] and [+16 .. ].
    const __bf16* Abase = Gbf + ((size_t)k * Nc + m0 + col) * (size_t)Nc + half * 8;
    // B (XcT) lane column d = t*16+col; K chunks: [j0+half*8 .. +7] and [+16 .. ].
    const __bf16* Bbase = XcT + ((size_t)b * DINP + col) * (size_t)Nc + half * 8;

    v8f acc[NB];
#pragma unroll
    for (int tI = 0; tI < NB; ++tI)
#pragma unroll
        for (int e = 0; e < 8; ++e) acc[tI][e] = 0.f;

    for (int j0 = 0; j0 < Nc; j0 += 32) {
        FragAB A;
        A.q[0] = *(const u32x4*)(Abase + j0);
        A.q[1] = *(const u32x4*)(Abase + j0 + 16);
        __builtin_prefetch(Abase + j0 + 256, 0, 1);   // stream-ahead on G
#pragma unroll
        for (int tI = 0; tI < NB; ++tI) {
            FragAB Bv;
            const __bf16* bp = Bbase + (size_t)tI * 16 * Nc + j0;
            Bv.q[0] = *(const u32x4*)(bp);
            Bv.q[1] = *(const u32x4*)(bp + 16);
            acc[tI] = __builtin_amdgcn_wmma_f32_16x16x32_bf16(
                false, A.v, false, Bv.v, (short)0, acc[tI], false, false);
        }
    }
#pragma unroll
    for (int tI = 0; tI < NB; ++tI)
#pragma unroll
        for (int r = 0; r < 8; ++r) {
            int m = m0 + r + 8 * half;
            supB[((size_t)b * Nc + m) * KSUP + k * DINP + tI * 16 + col] = (__bf16)acc[tI][r];
        }
}

// ----------- GEMM2 + fused LSTM: conv = sup @ W + b; gate update -----------
// Wave = 16 rows x gate-columns {h0, h0+64, h0+128, h0+192}+[0,16): all four
// gates for 16 h-channels -> LSTM update fused in the epilogue.

template <int KSUP>
__global__ __launch_bounds__(256) void k_gemm2_lstm(const __bf16* __restrict__ supB,
                                                    const __bf16* __restrict__ WT,
                                                    const float* __restrict__ bias,
                                                    float* __restrict__ hstate,
                                                    float* __restrict__ cstate,
                                                    float* __restrict__ hsave) {
    int b    = blockIdx.y;
    int wave = threadIdx.x >> 5;
    int lane = threadIdx.x & 31;
    int half = lane >> 4, col = lane & 15;
    int rowg = wave >> 2, hg = wave & 3;
    int m0    = blockIdx.x * 32 + rowg * 16;
    int h0col = hg * 16;

    const __bf16* Abase = supB + ((size_t)b * Nc + m0 + col) * KSUP + half * 8;
    const __bf16* Bbase = WT + ((size_t)b * DOUTc + h0col + col) * KSUP + half * 8;

    v8f acc[4];
#pragma unroll
    for (int g = 0; g < 4; ++g)
#pragma unroll
        for (int e = 0; e < 8; ++e) acc[g][e] = 0.f;

    for (int kd0 = 0; kd0 < KSUP; kd0 += 32) {
        FragAB A;
        A.q[0] = *(const u32x4*)(Abase + kd0);
        A.q[1] = *(const u32x4*)(Abase + kd0 + 16);
#pragma unroll
        for (int g = 0; g < 4; ++g) {
            FragAB Bv;
            const __bf16* bp = Bbase + (size_t)g * 64 * KSUP + kd0;
            Bv.q[0] = *(const u32x4*)(bp);
            Bv.q[1] = *(const u32x4*)(bp + 16);
            acc[g] = __builtin_amdgcn_wmma_f32_16x16x32_bf16(
                false, A.v, false, Bv.v, (short)0, acc[g], false, false);
        }
    }

    int hidx = h0col + col;
    float bgi = bias[b * DOUTc + hidx];
    float bgf = bias[b * DOUTc + 64 + hidx];
    float bgo = bias[b * DOUTc + 128 + hidx];
    float bgg = bias[b * DOUTc + 192 + hidx];
#pragma unroll
    for (int r = 0; r < 8; ++r) {
        int m = m0 + r + 8 * half;
        size_t si = ((size_t)b * Nc + m) * Hc + hidx;
        float gi = sigf(acc[0][r] + bgi);
        float gf = sigf(acc[1][r] + bgf);
        float go = sigf(acc[2][r] + bgo);
        float gg = tanhfast(acc[3][r] + bgg);
        float cn = gf * cstate[si] + gi * gg;
        float hn = go * tanhfast(cn);
        cstate[si] = cn;
        hstate[si] = hn;
        if (hsave) hsave[si] = hn;
    }
}

// --------------------------------- launcher --------------------------------

extern "C" void kernel_launch(void* const* d_in, const int* in_sizes, int n_in,
                              void* d_out, int out_size, void* d_ws, size_t ws_size,
                              hipStream_t stream) {
    (void)in_sizes; (void)n_in; (void)out_size; (void)ws_size;
    const float* G      = (const float*)d_in[0];
    const float* x_seq  = (const float*)d_in[1];
    const float* init_h = (const float*)d_in[2];
    const float* init_c = (const float*)d_in[3];
    const float* x_meta = (const float*)d_in[4];
    const float* p0[8];
    const float* p1[8];
    for (int i = 0; i < 8; ++i) { p0[i] = (const float*)d_in[5 + i]; p1[i] = (const float*)d_in[13 + i]; }

    char* ws = (char*)d_ws;
    size_t off = 0;
    auto take = [&](size_t bytes) -> char* {
        char* p = ws + off;
        off += (bytes + 255) & ~(size_t)255;
        return p;
    };
    __bf16* Gbf  = (__bf16*)take((size_t)Kc * Nc * Nc * 2);
    __bf16* XcT  = (__bf16*)take((size_t)Bc * DINP1 * Nc * 2);      // zeroed region start
    __bf16* supB = (__bf16*)take((size_t)Bc * Nc * KSUP1 * 2);
    __bf16* WT   = (__bf16*)take((size_t)Bc * DOUTc * KSUP1 * 2);
    float*  bias = (float*)take((size_t)Bc * DOUTc * 4);            // zeroed region end
    float*  hs0  = (float*)take((size_t)Tc * Bc * Nc * Hc * 4);
    float*  h0   = (float*)take((size_t)Bc * Nc * Hc * 4);
    float*  c0   = (float*)take((size_t)Bc * Nc * Hc * 4);
    float*  h1   = (float*)take((size_t)Bc * Nc * Hc * 4);
    float*  c1   = (float*)take((size_t)Bc * Nc * Hc * 4);

    const long bnh = (long)Bc * Nc * Hc;

    // One-time prep: G -> bf16, zero padded bf16 scratch, load init states.
    k_cvt_bf16<<<1024, 256, 0, stream>>>(G, Gbf, (long)Kc * Nc * Nc);
    size_t zbytes = (size_t)((char*)bias - (char*)XcT);
    k_zero32<<<512, 256, 0, stream>>>((unsigned int*)XcT, (long)(zbytes / 4));
    k_copy_f32<<<256, 256, 0, stream>>>(init_h, h0, bnh);
    k_copy_f32<<<256, 256, 0, stream>>>(init_h + bnh, h1, bnh);
    k_copy_f32<<<256, 256, 0, stream>>>(init_c, c0, bnh);
    k_copy_f32<<<256, 256, 0, stream>>>(init_c + bnh, c1, bnh);

    // -------- layer 0 over all timesteps (stores h into hs0) --------
    for (int t = 0; t < Tc; ++t) {
        k_xcat0<<<(Bc * Nc + 255) / 256, 256, 0, stream>>>(x_seq, h0, XcT, t);
        k_meta<<<dim3(Bc, 32), 256, 0, stream>>>(x_meta, t,
            p0[0], p0[1], p0[2], p0[3], p0[4], p0[5], p0[6], p0[7],
            WT, bias, DIN0, DINP0, KSUP0);
        k_gemm1<DINP0, KSUP0><<<dim3(Nc / 128, Kc, Bc), 256, 0, stream>>>(Gbf, XcT, supB);
        k_gemm2_lstm<KSUP0><<<dim3(Nc / 32, Bc), 256, 0, stream>>>(
            supB, WT, bias, h0, c0, hs0 + (size_t)t * bnh);
    }
    // -------- layer 1 over all timesteps (consumes hs0) --------
    for (int t = 0; t < Tc; ++t) {
        k_xcat1<<<(Bc * Nc + 255) / 256, 256, 0, stream>>>(hs0 + (size_t)t * bnh, h1, XcT);
        k_meta<<<dim3(Bc, 32), 256, 0, stream>>>(x_meta, t,
            p1[0], p1[1], p1[2], p1[3], p1[4], p1[5], p1[6], p1[7],
            WT, bias, DIN1, DINP1, KSUP1);
        k_gemm1<DINP1, KSUP1><<<dim3(Nc / 128, Kc, Bc), 256, 0, stream>>>(Gbf, XcT, supB);
        k_gemm2_lstm<KSUP1><<<dim3(Nc / 32, Bc), 256, 0, stream>>>(
            supB, WT, bias, h1, c1, (float*)nullptr);
    }

    // Output: [2][L][B][N][H] = {h0T, h1T, c0T, c1T}
    float* out = (float*)d_out;
    k_copy_f32<<<256, 256, 0, stream>>>(h0, out, bnh);
    k_copy_f32<<<256, 256, 0, stream>>>(h1, out + bnh, bnh);
    k_copy_f32<<<256, 256, 0, stream>>>(c0, out + 2 * bnh, bnh);
    k_copy_f32<<<256, 256, 0, stream>>>(c1, out + 3 * bnh, bnh);
}
